// ConvDVSGestureSNN_15831249453653
// MI455X (gfx1250) — matrix operand order repaired and checked by reading, not compiled, not run on hardware
//
#include <hip/hip_runtime.h>
#include <hip/hip_bf16.h>
#include <stdint.h>

// ---------------------------------------------------------------------------
// CDNA5 (gfx1250, wave32) spiking CNN, per-timestep kernel chain.
// conv2 = 25 shifted K=32 GEMMs, fc1/fc_out = GEMMs on v_wmma_f32_16x16x32_f16.
// Spike staging uses global_load_async_to_lds_b128 (ASYNCcnt); fc_out weights
// staged by the Tensor Data Mover (6-arg clang-23 builtin, TENSORcnt).
// ---------------------------------------------------------------------------

typedef __attribute__((ext_vector_type(16))) _Float16 v16h;
typedef __attribute__((ext_vector_type(8)))  _Float16 v8h;
typedef __attribute__((ext_vector_type(8)))  float    v8f;
typedef __attribute__((ext_vector_type(4)))  unsigned int u32x4;
typedef __attribute__((ext_vector_type(8)))  int      i32x8;
typedef __attribute__((ext_vector_type(4)))  int      i32x4;

union HalfVec16 { v16h v; v8h h[2]; };

#if defined(__has_builtin)
#  if __has_builtin(__builtin_amdgcn_tensor_load_to_lds)
#    define HAVE_TDM 1
#  endif
#endif
#ifndef HAVE_TDM
#  define HAVE_TDM 0
#endif

__device__ __forceinline__ float sigm(float x) { return 1.f / (1.f + __expf(-x)); }

#define B_    128
#define T_    50
#define NFC   256
#define NOUT  11

// ---- workspace layout (element offsets) -----------------------------------
// float region (zero-initialized each call)
#define O_VC1   0            // v_c1   : 128*32*14*14
#define O_VC2   802816       // v_c2   : 128*64*5*5
#define O_VFC   1007616      // v_fc   : 128*256
#define O_AFC   1040384      // a_fc   : 128*256
#define O_VOUT  1073152      // v_out  : 128*11
#define O_OSUM  1074560      // out_sum: 128*11
#define FLOATS_TOTAL 1075968
// half region (starts right after floats; byte offset 4303872, 64B aligned)
#define H_SPKFC 0            // spk_fc : 128*256        (zero-initialized)
#define H_SPK1  32768        // spk_c1 : 128*196*32     ([b][pix][ch], ch-minor)
#define H_SPK2  835584       // spk_c2 : 128*1600       ([b][k], k=oc*25+pp)
#define H_W2    1040384      // conv2 w, A-fragment swizzled: [khkw][mtile][lane][16]
#define H_FC1   1091584      // fc1_w fp16 [256][1600]
#define H_FCOUT 1501184      // fc_out_w fp16 padded [16][256]
#define HALFS_TOTAL 1505280
#define ZERO_DWORDS (FLOATS_TOTAL + 32768/2)   // floats + spk_fc halves, contiguous
// required ws_size: 4*FLOATS_TOTAL + 2*HALFS_TOTAL = 7,314,432 bytes (~7.3 MB)

// ---------------------------------------------------------------------------
// init / weight-conversion kernels
// ---------------------------------------------------------------------------
__global__ void k_zero(uint32_t* p, int n) {
    int i = blockIdx.x * blockDim.x + threadIdx.x;
    if (i < n) p[i] = 0u;
}

// conv2_w (64,32,5,5) f32 -> fp16, swizzled into WMMA A-fragment order:
// consumer reads v16h at index ((khkw*4 + mtile)*32 + lane).
// Contiguous K order used here (B matches it): lane holds row M=lane&15; halves h[0..15] = K kpart..kpart+15,
// kpart = (lane<16)?0:16.   (K axis = input channel)
__global__ void k_prep_w2(const float* __restrict__ w2, _Float16* __restrict__ w2h) {
    int i = blockIdx.x * blockDim.x + threadIdx.x;
    if (i >= 25 * 4 * 32 * 16) return;
    int h    = i & 15;
    int lane = (i >> 4) & 31;
    int m    = (i >> 9) & 3;
    int khkw = i >> 11;
    int oc    = m * 16 + (lane & 15);
    int kpart = (lane < 16) ? 0 : 16;
    int ic    = kpart + h;
    w2h[i] = (_Float16)w2[(oc * 32 + ic) * 25 + khkw];
}

__global__ void k_prep_fc1(const float* __restrict__ w, _Float16* __restrict__ wh) {
    int i = blockIdx.x * blockDim.x + threadIdx.x;
    if (i < NFC * 1600) wh[i] = (_Float16)w[i];
}

__global__ void k_prep_fcout(const float* __restrict__ w, _Float16* __restrict__ wh) {
    int i = blockIdx.x * blockDim.x + threadIdx.x;
    if (i >= 16 * 256) return;
    int n = i >> 8, k = i & 255;
    wh[i] = (_Float16)((n < NOUT) ? w[n * 256 + k] : 0.f);
}

// ---------------------------------------------------------------------------
// conv1 + BN-fold + pool + LIF1  (VALU; K=50 is too small for WMMA)
// ---------------------------------------------------------------------------
__global__ void __launch_bounds__(256)
k_conv1_lif(const float* __restrict__ x, const float* __restrict__ w1,
            const float* __restrict__ g1, const float* __restrict__ be1,
            const float* __restrict__ mu1, const float* __restrict__ var1,
            const float* __restrict__ bc1raw,
            float* __restrict__ v_c1, _Float16* __restrict__ spk1, int t) {
    int tid = blockIdx.x * blockDim.x + threadIdx.x;     // exact: 3136*256
    int b   = tid / (32 * 196);
    int r   = tid % (32 * 196);
    int oc  = r / 196;
    int pix = r % 196;
    int ph = pix / 14, pw = pix % 14;
    int y0 = 2 * ph, x0 = 2 * pw;
    const float* xb = x + (((size_t)b * T_ + t) * 2 * 32 * 32);
    float acc = 0.f;
#pragma unroll
    for (int ic = 0; ic < 2; ++ic) {
        const float* xc = xb + ic * 1024;
        const float* wc = w1 + (oc * 2 + ic) * 25;
#pragma unroll
        for (int ky = 0; ky < 5; ++ky) {
#pragma unroll
            for (int kx = 0; kx < 5; ++kx) {
                const float* p0 = xc + (y0 + ky) * 32 + (x0 + kx);
                acc += wc[ky * 5 + kx] * (p0[0] + p0[1] + p0[32] + p0[33]);
            }
        }
    }
    float s1   = g1[oc] * rsqrtf(var1[oc] + 1e-5f);
    float c1   = 0.25f * acc * s1 + (be1[oc] - mu1[oc] * s1);
    float beta = sigm(bc1raw[oc]);
    float v    = beta * v_c1[tid] + (1.f - beta) * c1;
    float s    = (v - 1.0f) > 0.f ? 1.f : 0.f;
    v_c1[tid]  = v * (1.f - s);
    spk1[(size_t)b * 6272 + pix * 32 + oc] = (_Float16)s;   // [b][pix][ch]
}

// ---------------------------------------------------------------------------
// conv2 via WMMA: 25 shifted 64x32 @ 32x16 GEMM-accumulations per N-tile,
// software-pipelined (register double buffering of A/B fragments).
// Spike tile staged to LDS with global_load_async_to_lds_b128 (ASYNCcnt).
// ---------------------------------------------------------------------------
__global__ void __launch_bounds__(256)
k_conv2_lif(const _Float16* __restrict__ spk1, const _Float16* __restrict__ w2h,
            const float* __restrict__ g2, const float* __restrict__ be2,
            const float* __restrict__ mu2, const float* __restrict__ var2,
            const float* __restrict__ bc2raw,
            float* __restrict__ v_c2, _Float16* __restrict__ spk2) {
    __shared__ __align__(16) _Float16 sSpk[196 * 32];   // [pix][ch], 12.25 KB
    __shared__ __align__(16) float    sOut[64 * 112];   // pre-pool conv output, 28 KB
    int b   = blockIdx.x;
    int tid = threadIdx.x;

    { // async-stage this image's spikes into LDS: 784 x B128 transfers
        const char* src = (const char*)(spk1 + (size_t)b * 6272);
        uint32_t ldsbase = (uint32_t)(size_t)sSpk;       // flat low 32 bits = LDS offset
        for (int i = tid; i < 784; i += 256) {
            uint32_t la = ldsbase + i * 16;
            uint64_t ga = (uint64_t)(size_t)(src + i * 16);
            asm volatile("global_load_async_to_lds_b128 %0, %1, off"
                         :: "v"(la), "v"(ga) : "memory");
        }
        asm volatile("s_wait_asynccnt 0x0" ::: "memory");
    }
    __syncthreads();

    int wave = tid >> 5, lane = tid & 31;
    if (wave < 7) {                          // wave-uniform branch; EXEC all-ones inside
        int  r  = lane & 15;
        bool hi = lane >= 16;
        int  p  = wave * 16 + r;             // pre-pool pixel this lane's B-column owns
        int  pc = (p < 100) ? p : 99;        // clamp padding lanes (result unused)
        int  oh = pc / 10, ow = pc % 10;
        const v16h* wptr = (const v16h*)w2h;
        const int bofs = hi ? 16 : 0;

        v8f acc0 = {}, acc1 = {}, acc2 = {}, acc3 = {};
        // prologue: fragments for khkw = 0
        v16h Bf = *(const v16h*)(sSpk + (oh * 14 + ow) * 32 + bofs);
        v16h A0 = wptr[0 * 32 + lane];
        v16h A1 = wptr[1 * 32 + lane];
        v16h A2 = wptr[2 * 32 + lane];
        v16h A3 = wptr[3 * 32 + lane];
        for (int khkw = 0; khkw < 25; ++khkw) {
            // prefetch next iteration's fragments (redundant reload on last iter)
            int nk = (khkw < 24) ? khkw + 1 : khkw;
            int kh = nk / 5, kw = nk % 5;
            v16h Bn = *(const v16h*)(sSpk + ((oh + kh) * 14 + (ow + kw)) * 32 + bofs);
            v16h N0 = wptr[(nk * 4 + 0) * 32 + lane];
            v16h N1 = wptr[(nk * 4 + 1) * 32 + lane];
            v16h N2 = wptr[(nk * 4 + 2) * 32 + lane];
            v16h N3 = wptr[(nk * 4 + 3) * 32 + lane];
            acc0 = __builtin_amdgcn_wmma_f32_16x16x32_f16(false, A0, false, Bf, (short)0, acc0, false, false);
            acc1 = __builtin_amdgcn_wmma_f32_16x16x32_f16(false, A1, false, Bf, (short)0, acc1, false, false);
            acc2 = __builtin_amdgcn_wmma_f32_16x16x32_f16(false, A2, false, Bf, (short)0, acc2, false, false);
            acc3 = __builtin_amdgcn_wmma_f32_16x16x32_f16(false, A3, false, Bf, (short)0, acc3, false, false);
            Bf = Bn; A0 = N0; A1 = N1; A2 = N2; A3 = N3;
        }
        // C layout: VGPR j -> row M = j (+8 for lanes 16..31); column N = lane&15
        int mb = hi ? 8 : 0;
#pragma unroll
        for (int j = 0; j < 8; ++j) {
            sOut[( 0 + j + mb) * 112 + p] = acc0[j];
            sOut[(16 + j + mb) * 112 + p] = acc1[j];
            sOut[(32 + j + mb) * 112 + p] = acc2[j];
            sOut[(48 + j + mb) * 112 + p] = acc3[j];
        }
    }
    __syncthreads();

    // BN-fold + 2x2 avg-pool + LIF2; write spikes flattened as [b][oc*25+pp]
    for (int i = tid; i < 64 * 25; i += 256) {
        int oc = i / 25, pp = i % 25;
        int ph = pp / 5, pw = pp % 5;
        const float* S = sOut + oc * 112;
        float c = 0.25f * (S[(2 * ph) * 10 + 2 * pw]     + S[(2 * ph) * 10 + 2 * pw + 1] +
                           S[(2 * ph + 1) * 10 + 2 * pw] + S[(2 * ph + 1) * 10 + 2 * pw + 1]);
        float s2 = g2[oc] * rsqrtf(var2[oc] + 1e-5f);
        c = c * s2 + (be2[oc] - mu2[oc] * s2);
        float beta = sigm(bc2raw[oc]);
        size_t idx = (size_t)b * 1600 + i;
        float v = beta * v_c2[idx] + (1.f - beta) * c;
        float s = (v - 1.f) > 0.f ? 1.f : 0.f;
        v_c2[idx] = v * (1.f - s);
        spk2[idx] = (_Float16)s;
    }
}

// ---------------------------------------------------------------------------
// fc1 (128x1600 @ 1600x256) via WMMA + adaptive LIF on the C fragment.
// 16 blocks * 8 waves = 128 waves; wave g owns (mtile = g>>4, ntile = g&15).
// K-loop register double-buffered.
// ---------------------------------------------------------------------------
__global__ void __launch_bounds__(256)
k_fc1_lif(const _Float16* __restrict__ spk2, const _Float16* __restrict__ fc1h,
          _Float16* __restrict__ spkfc, float* __restrict__ v_fc, float* __restrict__ a_fc,
          const float* __restrict__ alpha_raw, const float* __restrict__ rho_raw,
          const float* __restrict__ beta_a) {
    int wave = threadIdx.x >> 5, lane = threadIdx.x & 31;
    int g = blockIdx.x * 8 + wave;
    int mtile = g >> 4, ntile = g & 15;
    int  r  = lane & 15;
    bool hi = lane >= 16;
    int b = mtile * 16 + r;
    int n = ntile * 16 + r;
    v8f acc = {};
    const _Float16* arow = spk2 + (size_t)b * 1600 + (hi ? 16 : 0);  // A K-pattern
    const _Float16* brow = fc1h + (size_t)n * 1600 + (hi ? 16 : 0);  // B K-pattern
    HalfVec16 A;  v16h Bf;
    A.h[0] = *(const v8h*)(arow);
    A.h[1] = *(const v8h*)(arow + 8);
    Bf     = *(const v16h*)(brow);
    for (int ks = 0; ks < 50; ++ks) {
        int kn = ((ks < 49) ? ks + 1 : ks) * 32;
        __builtin_prefetch(brow + kn + 128, 0, 3);
        HalfVec16 An;  v16h Bn;
        An.h[0] = *(const v8h*)(arow + kn);
        An.h[1] = *(const v8h*)(arow + kn + 8);
        Bn      = *(const v16h*)(brow + kn);
        acc = __builtin_amdgcn_wmma_f32_16x16x32_f16(false, A.v, false, Bf, (short)0, acc, false, false);
        A = An; Bf = Bn;
    }
    // adaptive LIF, elementwise on C fragment (each (b,n) owned by one lane)
    float alpha = sigm(alpha_raw[n]);
    float rho   = sigm(rho_raw[n]);
    float ba    = beta_a[n];
    int mb = hi ? 8 : 0;
#pragma unroll
    for (int j = 0; j < 8; ++j) {
        int bb = mtile * 16 + j + mb;
        size_t idx = (size_t)bb * NFC + n;
        float sp_prev = (float)spkfc[idx];
        float a_ = rho * a_fc[idx] + (1.f - rho) * sp_prev;
        float v  = alpha * v_fc[idx] + (1.f - alpha) * acc[j];
        float s  = (v - (1.0f + ba * a_)) > 0.f ? 1.f : 0.f;
        v_fc[idx]  = v * (1.f - s);
        a_fc[idx]  = a_;
        spkfc[idx] = (_Float16)s;
    }
}

// ---------------------------------------------------------------------------
// fc_out (128x256 @ 256x16-padded) via WMMA + leaky output integrator.
// One block, 8 waves (wave = batch M-tile). Padded fp16 weights (8 KB) are
// staged into LDS by the Tensor Data Mover (1-D tile D#).
// ---------------------------------------------------------------------------
__global__ void __launch_bounds__(256)
k_fcout(const _Float16* __restrict__ spkfc, const _Float16* __restrict__ fcouth,
        float* __restrict__ v_out, float* __restrict__ out_sum,
        const float* __restrict__ beta_out_raw) {
    __shared__ __align__(16) _Float16 sB[16 * 256];     // 8 KB
#if HAVE_TDM
    if (threadIdx.x < 32) {                  // one wave issues the TDM descriptor
        uint64_t ga = (uint64_t)(size_t)fcouth;
        u32x4 g0;
        g0[0] = 1u;                                      // count=1 (valid descriptor)
        g0[1] = (uint32_t)(size_t)sB;                    // lds_addr (bytes)
        g0[2] = (uint32_t)ga;                            // global_addr[31:0]
        g0[3] = (uint32_t)(ga >> 32) | (2u << 30);       // global_addr[56:32] | type=2
        // data_size=2B (code 1); tensor_dim0 = tile_dim0 = 4096 elems; tensor_dim1=1;
        // tensor_dim0_stride = 4096; 1-D tile (tile_dim1=0 unused).
        i32x8 g1 = { 0x00010000, 0x10000000, 0x00010000, 0x10000000, 0, 4096, 0, 0 };
        i32x4 gz4 = { 0, 0, 0, 0 };
        i32x8 gz8 = { 0, 0, 0, 0, 0, 0, 0, 0 };
        __builtin_amdgcn_tensor_load_to_lds(g0, g1, gz4, gz4, gz8, 0);
        __builtin_amdgcn_s_wait_tensorcnt(0);
    }
    __syncthreads();
#else
    // fallback: cooperative copy (2048 dwords)
    {
        const uint32_t* srcw = (const uint32_t*)fcouth;
        uint32_t* dstw = (uint32_t*)sB;
        for (int i = threadIdx.x; i < 2048; i += 256) dstw[i] = srcw[i];
    }
    __syncthreads();
#endif
    int wave = threadIdx.x >> 5, lane = threadIdx.x & 31;
    int  r  = lane & 15;
    bool hi = lane >= 16;
    int b = wave * 16 + r;
    int n = r;                                   // N-tile 0 (11 padded to 16)
    v8f acc = {};
    const _Float16* arow = spkfc + (size_t)b * 256 + (hi ? 16 : 0);
    const _Float16* brow = sB + n * 256 + (hi ? 16 : 0);    // LDS -> ds_load fragments
#pragma unroll
    for (int ks = 0; ks < 8; ++ks) {
        int k0 = ks * 32;
        HalfVec16 A;
        A.h[0] = *(const v8h*)(arow + k0);
        A.h[1] = *(const v8h*)(arow + k0 + 8);
        v16h Bf = *(const v16h*)(brow + k0);
        acc = __builtin_amdgcn_wmma_f32_16x16x32_f16(false, A.v, false, Bf, (short)0, acc, false, false);
    }
    float bo = sigm(beta_out_raw[0]);
    int mb = hi ? 8 : 0;
    if (n < NOUT) {
#pragma unroll
        for (int j = 0; j < 8; ++j) {
            int bb = wave * 16 + j + mb;
            int idx = bb * NOUT + n;
            float v = bo * v_out[idx] + (1.f - bo) * acc[j];
            v_out[idx] = v;
            out_sum[idx] += v;
        }
    }
}

__global__ void k_final(const float* __restrict__ out_sum, float* __restrict__ out, int n) {
    int i = blockIdx.x * blockDim.x + threadIdx.x;
    if (i < n) out[i] = out_sum[i] * (1.f / (float)T_);
}

// ---------------------------------------------------------------------------
extern "C" void kernel_launch(void* const* d_in, const int* in_sizes, int n_in,
                              void* d_out, int out_size, void* d_ws, size_t ws_size,
                              hipStream_t stream) {
    const float* x        = (const float*)d_in[0];
    const float* conv1_w  = (const float*)d_in[1];
    const float* bn1_g    = (const float*)d_in[2];
    const float* bn1_b    = (const float*)d_in[3];
    const float* bn1_m    = (const float*)d_in[4];
    const float* bn1_v    = (const float*)d_in[5];
    const float* conv2_w  = (const float*)d_in[6];
    const float* bn2_g    = (const float*)d_in[7];
    const float* bn2_b    = (const float*)d_in[8];
    const float* bn2_m    = (const float*)d_in[9];
    const float* bn2_v    = (const float*)d_in[10];
    const float* bc1raw   = (const float*)d_in[11];
    const float* bc2raw   = (const float*)d_in[12];
    const float* fc1_w    = (const float*)d_in[13];
    const float* alpha_r  = (const float*)d_in[14];
    const float* rho_r    = (const float*)d_in[15];
    const float* beta_a   = (const float*)d_in[16];
    const float* fcout_w  = (const float*)d_in[17];
    const float* beta_out = (const float*)d_in[18];
    float* out = (float*)d_out;

    float*    wsf = (float*)d_ws;
    _Float16* wsh = (_Float16*)(wsf + FLOATS_TOTAL);

    float* v_c1    = wsf + O_VC1;
    float* v_c2    = wsf + O_VC2;
    float* v_fc    = wsf + O_VFC;
    float* a_fc    = wsf + O_AFC;
    float* v_out   = wsf + O_VOUT;
    float* out_sum = wsf + O_OSUM;
    _Float16* spkfc  = wsh + H_SPKFC;
    _Float16* spk1   = wsh + H_SPK1;
    _Float16* spk2   = wsh + H_SPK2;
    _Float16* w2h    = wsh + H_W2;
    _Float16* fc1h   = wsh + H_FC1;
    _Float16* fcouth = wsh + H_FCOUT;

    // zero state (floats + spk_fc, contiguous dword span)
    k_zero<<<(ZERO_DWORDS + 255) / 256, 256, 0, stream>>>((uint32_t*)d_ws, ZERO_DWORDS);
    // fp16 weight conversion / swizzle (deterministic, cheap, L2-resident)
    k_prep_w2   <<<(25 * 4 * 32 * 16 + 255) / 256, 256, 0, stream>>>(conv2_w, w2h);
    k_prep_fc1  <<<(NFC * 1600 + 255) / 256, 256, 0, stream>>>(fc1_w, fc1h);
    k_prep_fcout<<<(16 * 256 + 255) / 256, 256, 0, stream>>>(fcout_w, fcouth);

    for (int t = 0; t < T_; ++t) {
        k_conv1_lif<<<3136, 256, 0, stream>>>(x, conv1_w, bn1_g, bn1_b, bn1_m, bn1_v,
                                              bc1raw, v_c1, spk1, t);
        k_conv2_lif<<<B_, 256, 0, stream>>>(spk1, w2h, bn2_g, bn2_b, bn2_m, bn2_v,
                                            bc2raw, v_c2, spk2);
        k_fc1_lif<<<16, 256, 0, stream>>>(spk2, fc1h, spkfc, v_fc, a_fc,
                                          alpha_r, rho_r, beta_a);
        k_fcout<<<1, 256, 0, stream>>>(spkfc, fcouth, v_out, out_sum, beta_out);
    }
    k_final<<<(B_ * NOUT + 255) / 256, 256, 0, stream>>>(out_sum, out, B_ * NOUT);
}
